// Decoder_50440095924703
// MI455X (gfx1250) — compile-verified, matrix operand order checked by reading, not run on hardware
//
#include <hip/hip_runtime.h>
#include <hip/hip_bf16.h>

typedef __attribute__((ext_vector_type(16))) __bf16 v16bf;
typedef __attribute__((ext_vector_type(8)))  float  v8f;

#define D_    512
#define B_    32
#define LX_   48
#define LY_   32
#define V_    32000
#define NGATE 2048     // 4*D
#define KCAT  2048     // DW + 2D + D = 512 + 1024 + 512

// K index inside a 16-bit WMMA A/B fragment: element e (0..15), lane group g (0..1)
__device__ __forceinline__ int kmap(int e, int g) {
  return (e < 8) ? (8 * g + e) : (8 + 8 * g + e);
}

// Load one A fragment row-chunk from contiguous fp32 memory (p = row + kt*32).
// Per lane: two contiguous 32B fp32 chunks -> 8x v_cvt_pk_bf16_f32, no branches.
__device__ __forceinline__ v16bf load_a_row(const float* __restrict__ p, int g) {
  v16bf a;
#pragma unroll
  for (int e = 0; e < 8; ++e) a[e] = (__bf16)p[8 * g + e];
#pragma unroll
  for (int e = 0; e < 8; ++e) a[8 + e] = (__bf16)p[16 + 8 * g + e];
  return a;
}

// One K-step of a 4-wide-N WMMA tile row: 4 B-fragment loads + 4 WMMAs.
__device__ __forceinline__ void wmma4(const v16bf a, const __bf16* __restrict__ Bf,
                                      int ntBase, int KT, int kt, int lane, v8f acc[4]) {
#pragma unroll
  for (int j = 0; j < 4; ++j) {
    const v16bf b = *(const v16bf*)(Bf + ((size_t)((ntBase + j) * KT + kt) * 32 + lane) * 16);
    acc[j] = __builtin_amdgcn_wmma_f32_16x16x32_bf16(false, a, false, b, (short)0, acc[j],
                                                     false, false);
  }
}

// ---------------------------------------------------------------------------
// Pack a row-major weight matrix (concat of W0[N,ksplit] | W1[N,K-ksplit]) into
// bf16 WMMA B-fragments: frag f = nt*KT + kt, 32 lanes x 16 bf16 contiguous.
// ---------------------------------------------------------------------------
__global__ void pack_w_kernel(const float* __restrict__ W0, const float* __restrict__ W1,
                              int ksplit, int K, int KT, __bf16* __restrict__ dst, int nfrag)
{
  int tid  = blockIdx.x * blockDim.x + threadIdx.x;
  int lane = tid & 31;
  int f    = tid >> 5;
  if (f >= nfrag) return;
  int kt = f % KT;
  int nt = f / KT;
  int n  = nt * 16 + (lane & 15);
  int g  = lane >> 4;
  int k1 = K - ksplit;
#pragma unroll
  for (int e = 0; e < 16; ++e) {
    int k = kt * 32 + kmap(e, g);
    float v = (k < ksplit) ? W0[(size_t)n * ksplit + k]
                           : W1[(size_t)n * k1 + (k - ksplit)];
    dst[(size_t)f * 512 + lane * 16 + e] = (__bf16)v;
  }
}

// Pack pre_readouts [1024, 512] f32 -> bf16 WMMA A-fragments (MT=64, KT=16)
__global__ void pack_pre_kernel(const float* __restrict__ pre, __bf16* __restrict__ dst)
{
  int tid  = blockIdx.x * blockDim.x + threadIdx.x;   // 32768
  int lane = tid & 31;
  int f    = tid >> 5;                                 // < 1024
  if (f >= 1024) return;
  int kt = f & 15, mt = f >> 4;
  int m  = mt * 16 + (lane & 15);
  int g  = lane >> 4;
  const float* p = pre + (size_t)m * D_ + kt * 32;
#pragma unroll
  for (int e = 0; e < 8; ++e) dst[(size_t)f * 512 + lane * 16 + e] = (__bf16)p[8 * g + e];
#pragma unroll
  for (int e = 0; e < 8; ++e) dst[(size_t)f * 512 + lane * 16 + 8 + e] = (__bf16)p[16 + 8 * g + e];
}

__global__ void init_state_kernel(const float* __restrict__ h0, const float* __restrict__ c0,
                                  float* __restrict__ h, float* __restrict__ c,
                                  float* __restrict__ feed)
{
  int i = blockIdx.x * blockDim.x + threadIdx.x;   // 32768 threads
  if (i < B_ * D_) { h[i] = h0[i]; c[i] = c0[i]; }
  if (i < B_ * 2 * D_) feed[i] = 0.f;              // input_feed init = zeros
}

// ---------------------------------------------------------------------------
// gates = [emb_t, feed, h] @ [W_ih | W_hh]^T + b_ih + b_hh
// M=32 (2 mt), N=2048 (32 n-groups of 4 tiles), K=2048 (64 kt). 64 waves.
// kt regions are uniform: [0,16)=emb, [16,48)=feed, [48,64)=h -> 3 branch-free loops.
// ---------------------------------------------------------------------------
__global__ void gemm_gates_kernel(const __bf16* __restrict__ Bfrag,
                                  const int* __restrict__ y_train, int t,
                                  const float* __restrict__ word_emb,
                                  const float* __restrict__ feed,
                                  const float* __restrict__ h,
                                  const float* __restrict__ b_ih,
                                  const float* __restrict__ b_hh,
                                  float* __restrict__ gates)
{
  const int lane = threadIdx.x & 31;
  const int flat = blockIdx.x * (blockDim.x >> 5) + (threadIdx.x >> 5);  // 64 waves
  const int mt = flat >> 5;
  const int ntBase = (flat & 31) * 4;
  const int ma = mt * 16 + (lane & 15);
  const int g  = lane >> 4;
  const int yb = y_train[ma * LY_ + t];
  const float* embp  = word_emb + (size_t)yb * D_;
  const float* feedp = feed + ma * 1024 - 512;     // so feedp[kt*32] is valid for kt>=16
  const float* hp    = h + ma * D_ - 1536;
  v8f acc[4] = {};
  for (int kt = 0; kt < 16; ++kt) {
    const v16bf a = load_a_row(embp + kt * 32, g);
    wmma4(a, Bfrag, ntBase, 64, kt, lane, acc);
  }
  for (int kt = 16; kt < 48; ++kt) {
    const v16bf a = load_a_row(feedp + kt * 32, g);
    wmma4(a, Bfrag, ntBase, 64, kt, lane, acc);
  }
  for (int kt = 48; kt < 64; ++kt) {
    const v16bf a = load_a_row(hp + kt * 32, g);
    wmma4(a, Bfrag, ntBase, 64, kt, lane, acc);
  }
#pragma unroll
  for (int j = 0; j < 4; ++j) {
    const int nc = (ntBase + j) * 16 + (lane & 15);
#pragma unroll
    for (int r = 0; r < 8; ++r) {
      const int mr = mt * 16 + r + 8 * g;
      gates[mr * NGATE + nc] = acc[j][r] + b_ih[nc] + b_hh[nc];
    }
  }
}

// LSTM pointwise: i,f,g,o gate order (torch LSTMCell). Updates h,c in place.
__global__ void lstm_kernel(const float* __restrict__ gates,
                            float* __restrict__ h, float* __restrict__ c)
{
  int idx = blockIdx.x * blockDim.x + threadIdx.x;
  if (idx >= B_ * D_) return;
  int b = idx >> 9, d = idx & 511;
  const float* gb = gates + b * NGATE;
  float gi = gb[d], gf = gb[512 + d], gg = gb[1024 + d], go = gb[1536 + d];
  float si = 1.f / (1.f + expf(-gi));
  float sf = 1.f / (1.f + expf(-gf));
  float so = 1.f / (1.f + expf(-go));
  float cn = sf * c[idx] + si * tanhf(gg);
  c[idx] = cn;
  h[idx] = so * tanhf(cn);
}

// q = h_new @ w_trg_W^T + w_trg_b.  M=32, N=512 (8 n-groups), K=512 (16 kt). 16 waves.
__global__ void gemm_q_kernel(const __bf16* __restrict__ Bfrag,
                              const float* __restrict__ h,
                              const float* __restrict__ w_trg_b,
                              float* __restrict__ q)
{
  const int lane = threadIdx.x & 31;
  const int flat = blockIdx.x * (blockDim.x >> 5) + (threadIdx.x >> 5);  // 16 waves
  const int mt = flat >> 3;
  const int ntBase = (flat & 7) * 4;
  const int ma = mt * 16 + (lane & 15);
  const int g  = lane >> 4;
  const float* hp = h + ma * D_;
  v8f acc[4] = {};
  for (int kt = 0; kt < 16; ++kt) {
    const v16bf a = load_a_row(hp + kt * 32, g);
    wmma4(a, Bfrag, ntBase, 16, kt, lane, acc);
  }
#pragma unroll
  for (int j = 0; j < 4; ++j) {
    const int nc = (ntBase + j) * 16 + (lane & 15);
#pragma unroll
    for (int r = 0; r < 8; ++r) {
      const int mr = mt * 16 + r + 8 * g;
      q[mr * D_ + nc] = acc[j][r] + w_trg_b[nc];
    }
  }
}

// MLP attention + softmax + context. One 256-thread block per batch element.
__global__ void attn_kernel(const float* __restrict__ x_enc,
                            const float* __restrict__ x_enc_k,
                            const unsigned char* __restrict__ x_mask,
                            const float* __restrict__ q,
                            const float* __restrict__ w_att_W,
                            const float* __restrict__ w_att_b,
                            float* __restrict__ ctx)
{
  __shared__ float s_q[D_];
  __shared__ float s_sc[LX_];
  __shared__ float s_inv;
  const int b = blockIdx.x, tid = threadIdx.x;
  s_q[tid]       = q[b * D_ + tid];
  s_q[tid + 256] = q[b * D_ + 256 + tid];
  __syncthreads();
  const int w = tid >> 5, lane = tid & 31;
  for (int l = w; l < LX_; l += 8) {
    const float* xk = x_enc_k + ((size_t)b * LX_ + l) * D_;
    float partial = 0.f;
    for (int d = lane; d < D_; d += 32)
      partial += w_att_W[d] * tanhf(xk[d] + s_q[d]);
#pragma unroll
    for (int off = 16; off; off >>= 1) partial += __shfl_xor(partial, off, 32);
    if (lane == 0) {
      float sc = partial + w_att_b[0];
      if (x_mask[b * LX_ + l]) sc = -1e9f;
      s_sc[l] = sc;
    }
  }
  __syncthreads();
  if (tid == 0) {
    float m = s_sc[0];
    for (int l = 1; l < LX_; ++l) m = fmaxf(m, s_sc[l]);
    float s = 0.f;
    for (int l = 0; l < LX_; ++l) { float e = expf(s_sc[l] - m); s_sc[l] = e; s += e; }
    s_inv = 1.f / s;
  }
  __syncthreads();
  for (int d = tid; d < 2 * D_; d += 256) {
    float acc = 0.f;
    for (int l = 0; l < LX_; ++l)
      acc += s_sc[l] * x_enc[((size_t)b * LX_ + l) * (2 * D_) + d];
    ctx[b * 2 * D_ + d] = acc * s_inv;   // ctx doubles as next step's input feed
  }
}

// pre = tanh([h_new, ctx] @ ctx2r_W^T).  M=32, N=512 (8 n-groups), K=1536 (48 kt). 16 waves.
// kt regions uniform: [0,16)=h, [16,48)=ctx -> 2 branch-free loops.
__global__ void gemm_pre_kernel(const __bf16* __restrict__ Bfrag,
                                const float* __restrict__ h,
                                const float* __restrict__ ctx,
                                int t, float* __restrict__ pre)
{
  const int lane = threadIdx.x & 31;
  const int flat = blockIdx.x * (blockDim.x >> 5) + (threadIdx.x >> 5);  // 16 waves
  const int mt = flat >> 3;
  const int ntBase = (flat & 7) * 4;
  const int ma = mt * 16 + (lane & 15);
  const int g  = lane >> 4;
  const float* hp = h + ma * D_;
  const float* cp = ctx + ma * 1024 - 512;         // cp[kt*32] valid for kt>=16
  v8f acc[4] = {};
  for (int kt = 0; kt < 16; ++kt) {
    const v16bf a = load_a_row(hp + kt * 32, g);
    wmma4(a, Bfrag, ntBase, 48, kt, lane, acc);
  }
  for (int kt = 16; kt < 48; ++kt) {
    const v16bf a = load_a_row(cp + kt * 32, g);
    wmma4(a, Bfrag, ntBase, 48, kt, lane, acc);
  }
#pragma unroll
  for (int j = 0; j < 4; ++j) {
    const int nc = (ntBase + j) * 16 + (lane & 15);
#pragma unroll
    for (int r = 0; r < 8; ++r) {
      const int mr = mt * 16 + r + 8 * g;          // batch row b
      pre[(size_t)(mr * LY_ + t) * D_ + nc] = tanhf(acc[j][r]);   // [B, Ly, D]
    }
  }
}

// logits = pre @ readout_W^T.  M=1024 (64 mt), N=32000 (500 n-groups of 4), K=512.
// bf16 weights (32.8 MB) are L2-resident; A fragment reused x4 per K-step.
__global__ void gemm_readout_kernel(const __bf16* __restrict__ Afrag,
                                    const __bf16* __restrict__ Bfrag,
                                    float* __restrict__ out)
{
  const int lane = threadIdx.x & 31;
  const int flat = blockIdx.x * (blockDim.x >> 5) + (threadIdx.x >> 5);  // 32000 waves
  const int ng = flat % 500;
  const int mt = flat / 500;
  const int ntBase = ng * 4;
  v8f acc[4] = {};
  for (int kt = 0; kt < 16; ++kt) {
    const v16bf a = *(const v16bf*)(Afrag + ((size_t)(mt * 16 + kt) * 32 + lane) * 16);
    wmma4(a, Bfrag, ntBase, 16, kt, lane, acc);
  }
  const int g = lane >> 4, nc = lane & 15;
#pragma unroll
  for (int j = 0; j < 4; ++j) {
    const int n = (ntBase + j) * 16 + nc;
#pragma unroll
    for (int r = 0; r < 8; ++r) {
      const size_t mr = (size_t)(mt * 16 + r + 8 * g);
      out[mr * V_ + n] = acc[j][r];
    }
  }
}

extern "C" void kernel_launch(void* const* d_in, const int* in_sizes, int n_in,
                              void* d_out, int out_size, void* d_ws, size_t ws_size,
                              hipStream_t stream)
{
  (void)in_sizes; (void)n_in; (void)out_size; (void)ws_size;
  const float* x_enc     = (const float*)d_in[0];
  const float* x_enc_k   = (const float*)d_in[1];
  const float* h0        = (const float*)d_in[2];
  const float* c0        = (const float*)d_in[3];
  const unsigned char* x_mask = (const unsigned char*)d_in[4];
  const int*   y_train   = (const int*)d_in[5];
  const float* word_emb  = (const float*)d_in[6];
  const float* W_ih      = (const float*)d_in[7];
  const float* W_hh      = (const float*)d_in[8];
  const float* b_ih      = (const float*)d_in[9];
  const float* b_hh      = (const float*)d_in[10];
  const float* w_trg_W   = (const float*)d_in[11];
  const float* w_trg_b   = (const float*)d_in[12];
  const float* w_att_W   = (const float*)d_in[13];
  const float* w_att_b   = (const float*)d_in[14];
  const float* ctx2r_W   = (const float*)d_in[15];
  const float* readout_W = (const float*)d_in[16];
  float* out = (float*)d_out;

  // Workspace carve-out (256B aligned)
  char* p = (char*)d_ws;
  auto take = [&](size_t bytes) -> char* {
    char* r = p; p += (bytes + 255) & ~(size_t)255; return r;
  };
  __bf16* wcat_f = (__bf16*)take((size_t)128 * 64 * 512 * 2);   // [W_ih|W_hh] frags, 8 MB
  __bf16* wtrg_f = (__bf16*)take((size_t)32 * 16 * 512 * 2);    // w_trg frags
  __bf16* c2r_f  = (__bf16*)take((size_t)32 * 48 * 512 * 2);    // ctx2r frags
  __bf16* rd_f   = (__bf16*)take((size_t)2000 * 16 * 512 * 2);  // readout frags, 32.8 MB
  float*  gates  = (float*)take((size_t)B_ * NGATE * 4);
  float*  h      = (float*)take((size_t)B_ * D_ * 4);
  float*  c      = (float*)take((size_t)B_ * D_ * 4);
  float*  q      = (float*)take((size_t)B_ * D_ * 4);
  float*  feed   = (float*)take((size_t)B_ * 2 * D_ * 4);       // ctx / input_feed
  float*  pre    = (float*)take((size_t)B_ * LY_ * D_ * 4);     // [B, Ly, D]
  __bf16* pre_f  = (__bf16*)take((size_t)64 * 16 * 512 * 2);    // pre A-frags

  // One-time (per call) weight packing fp32 -> bf16 fragment layout
  pack_w_kernel<<<(128 * 64 * 32) / 256, 256, 0, stream>>>(W_ih, W_hh, 1536, KCAT, 64, wcat_f, 128 * 64);
  pack_w_kernel<<<(32 * 16 * 32) / 256, 256, 0, stream>>>(w_trg_W, w_trg_W, 512, 512, 16, wtrg_f, 32 * 16);
  pack_w_kernel<<<(32 * 48 * 32) / 256, 256, 0, stream>>>(ctx2r_W, ctx2r_W, 1536, 1536, 48, c2r_f, 32 * 48);
  pack_w_kernel<<<(2000 * 16 * 32) / 256, 256, 0, stream>>>(readout_W, readout_W, 512, 512, 16, rd_f, 2000 * 16);
  init_state_kernel<<<128, 256, 0, stream>>>(h0, c0, h, c, feed);

  // Sequential decode: 32 steps, stream-ordered dependence through h/c/feed
  for (int t = 0; t < LY_; ++t) {
    gemm_gates_kernel<<<8, 256, 0, stream>>>(wcat_f, y_train, t, word_emb, feed, h, b_ih, b_hh, gates);
    lstm_kernel<<<64, 256, 0, stream>>>(gates, h, c);
    gemm_q_kernel<<<2, 256, 0, stream>>>(wtrg_f, h, w_trg_b, q);
    attn_kernel<<<B_, 256, 0, stream>>>(x_enc, x_enc_k, x_mask, q, w_att_W, w_att_b, feed);
    gemm_pre_kernel<<<2, 256, 0, stream>>>(c2r_f, h, feed, t, pre);
  }

  // Readout: [1024, 512] x [512, 32000] -> out [B, Ly, V]
  pack_pre_kernel<<<128, 256, 0, stream>>>(pre, pre_f);
  gemm_readout_kernel<<<4000, 256, 0, stream>>>(pre_f, rd_f, out);
}
